// PAPF_78073915507322
// MI455X (gfx1250) — matrix operand order repaired and checked by reading, not compiled, not run on hardware
//
#include <hip/hip_runtime.h>
#include <cstdint>
#include <cstddef>

// ---------------------------------------------------------------------------
// Types for CDNA5 WMMA (wave32): V_WMMA_F32_16X16X32_F16
// ---------------------------------------------------------------------------
typedef _Float16 half_t;
typedef __attribute__((ext_vector_type(8)))  _Float16 v8h;
typedef __attribute__((ext_vector_type(16))) _Float16 v16h;
typedef __attribute__((ext_vector_type(8)))  float    v8f;

#define B_    256
#define PAST_ 512
#define PRED_ 48
#define H_    256
#define G_    768   // 3*H
#define BCH   16    // batch rows per workgroup in GRU scan

__device__ __forceinline__ float sigmoidf_(float x) { return 1.0f / (1.0f + __expf(-x)); }

// ---------------------------------------------------------------------------
// Weight packing: dst (768 x (din+256)) f16, row n = [Wih[n,:din] | Whh[n,:256]]
// ---------------------------------------------------------------------------
__global__ void gruw_pack_kernel(const float* __restrict__ Wih,
                                 const float* __restrict__ Whh,
                                 half_t* __restrict__ dst, int din) {
  int K = din + H_;
  size_t i = (size_t)blockIdx.x * 256 + threadIdx.x;
  if (i >= (size_t)G_ * K) return;
  int row = (int)(i / K), k = (int)(i % K);
  float v = (k < din) ? Wih[(size_t)row * din + k] : Whh[(size_t)row * H_ + (k - din)];
  dst[i] = (half_t)v;
}

__global__ void cast_f16_kernel(const float* __restrict__ src, half_t* __restrict__ dst, int n) {
  int i = blockIdx.x * 256 + threadIdx.x;
  if (i < n) dst[i] = (half_t)src[i];
}

// split attn_W (256 x 768) into Wq (256x256) and We (256x512), f16
__global__ void attnw_split_kernel(const float* __restrict__ attnW,
                                   half_t* __restrict__ Wq, half_t* __restrict__ We) {
  size_t i = (size_t)blockIdx.x * 256 + threadIdx.x;  // over 256*768
  int row = (int)(i / G_), k = (int)(i % G_);
  float v = attnW[i];
  if (k < H_) Wq[(size_t)row * H_ + k] = (half_t)v;
  else        We[(size_t)row * 512 + (k - H_)] = (half_t)v;
}

// encoder input FC (NX=1): Xbuf[bt, h] = relu(x[bt]*W[h]+b[h])  (f16)
__global__ void encfc_kernel(const float* __restrict__ x, const float* __restrict__ W,
                             const float* __restrict__ bias, half_t* __restrict__ Xbuf) {
  size_t idx = (size_t)blockIdx.x * 256 + threadIdx.x;  // B*T*256 total
  int h = (int)(idx & 255); size_t bt = idx >> 8;
  Xbuf[idx] = (half_t)fmaxf(x[bt] * W[h] + bias[h], 0.0f);
}

__global__ void y0_kernel(const float* __restrict__ x_past, float* __restrict__ yprev) {
  int b = threadIdx.x;
  if (b < B_) yprev[b] = x_past[(size_t)b * PAST_ + (PAST_ - 1)];
}

// ---------------------------------------------------------------------------
// GRU scan kernel. 512 threads = 16 waves. BCH=16 batch rows per block.
// Wave w owns gate tiles {w (r), w+16 (z), w+32 (n)} -> same 16 h-columns, so
// the full gate nonlinearity is register resident. h lives in LDS across the
// sequential T loop; only workgroup barriers between timesteps.
// K-loop is split in two straight-line loops (x-part then h-part) so each has
// a dedicated accumulator bank: no per-iteration predicate, no acc copies.
// bidir: blockIdx.y selects direction (forward/backward weights).
// ---------------------------------------------------------------------------
__global__ __launch_bounds__(512) void gru_scan_kernel(
    const half_t* __restrict__ X, int din,                 // X: (B, Tsteps, din) f16
    const half_t* __restrict__ Wf, const half_t* __restrict__ Wb,   // 768 x (din+256)
    const float* __restrict__ bihf, const float* __restrict__ bihb,
    const float* __restrict__ bhhf, const float* __restrict__ bhhb,
    const float* __restrict__ h_in,                        // nullable; (B,256) f32
    float* __restrict__ h_out32,                           // nullable; (B,256) f32
    half_t* __restrict__ h_out16, int hostride,            // nullable; final h f16
    half_t* __restrict__ out16, int ostride,               // per-step output f16
    int Tsteps, int bidir) {
  __shared__ half_t Xs[BCH * 1024];
  __shared__ float  Hs32[BCH * H_];
  __shared__ half_t Hs16[BCH * H_];

  const int tid  = threadIdx.x;
  const int wave = tid >> 5;
  const int lane = tid & 31;
  const int dir  = bidir ? (int)blockIdx.y : 0;
  const int bbase = blockIdx.x * BCH;
  const int K = din + H_;
  const int ocoff = bidir ? dir * H_ : 0;

  const half_t* Wcat = dir ? Wb : Wf;
  const float*  bih  = dir ? bihb : bihf;
  const float*  bhh  = dir ? bhhb : bhhf;

  for (int i = tid; i < BCH * H_; i += 512) {
    float hv = h_in ? h_in[(size_t)(bbase + i / H_) * H_ + (i % H_)] : 0.0f;
    Hs32[i] = hv; Hs16[i] = (half_t)hv;
  }

  const int c  = lane & 15;
  const int hi = lane >> 4;
  const int colR = wave * 16 + c;          // 0..255
  const half_t* WrowR = Wcat + (size_t)(colR)            * K;
  const half_t* WrowZ = Wcat + (size_t)(H_ + colR)       * K;
  const half_t* WrowN = Wcat + (size_t)(2 * H_ + colR)   * K;
  const float bihR = bih[colR], bihZ = bih[H_ + colR], bihN = bih[2 * H_ + colR];
  const float bhhR = bhh[colR], bhhZ = bhh[H_ + colR], bhhN = bhh[2 * H_ + colR];
  const int rb = hi * 8;

  for (int t = 0; t < Tsteps; ++t) {
    const int tact = dir ? (Tsteps - 1 - t) : t;
    __syncthreads();
    // cooperative load x_t tile (16 x din) into LDS
    const int nvec = BCH * din / 8;
    for (int i = tid; i < nvec; i += 512) {
      int m = i / (din / 8), kk = (i % (din / 8)) * 8;
      *(v8h*)&Xs[m * din + kk] =
          *(const v8h*)&X[((size_t)(bbase + m) * Tsteps + tact) * din + kk];
    }
    __syncthreads();

    // ---- gi = x @ Wih^T : straight-line K loop over the x range ----
    v8f aR = {}, aZ = {}, aN = {};
    {
      const half_t* arow = Xs + c * din;
      for (int k0 = 0; k0 < din; k0 += 32) {
        v16h a;
        ((v8h*)&a)[0] = *(const v8h*)(arow + k0 + hi * 16);
        ((v8h*)&a)[1] = *(const v8h*)(arow + k0 + hi * 16 + 8);
        v16h bRv = *(const v16h*)(WrowR + k0 + hi * 16);
        v16h bZv = *(const v16h*)(WrowZ + k0 + hi * 16);
        v16h bNv = *(const v16h*)(WrowN + k0 + hi * 16);
        aR = __builtin_amdgcn_wmma_f32_16x16x32_f16(false, a, false, bRv, (short)0, aR, false, false);
        aZ = __builtin_amdgcn_wmma_f32_16x16x32_f16(false, a, false, bZv, (short)0, aZ, false, false);
        aN = __builtin_amdgcn_wmma_f32_16x16x32_f16(false, a, false, bNv, (short)0, aN, false, false);
      }
    }
    // ---- gh = h @ Whh^T : straight-line K loop over the h range ----
    v8f hR = {}, hZ = {}, hN = {};
    {
      const half_t* arow = Hs16 + c * H_;
      const half_t* wR = WrowR + din;
      const half_t* wZ = WrowZ + din;
      const half_t* wN = WrowN + din;
      for (int k0 = 0; k0 < H_; k0 += 32) {
        v16h a;
        ((v8h*)&a)[0] = *(const v8h*)(arow + k0 + hi * 16);
        ((v8h*)&a)[1] = *(const v8h*)(arow + k0 + hi * 16 + 8);
        v16h bRv = *(const v16h*)(wR + k0 + hi * 16);
        v16h bZv = *(const v16h*)(wZ + k0 + hi * 16);
        v16h bNv = *(const v16h*)(wN + k0 + hi * 16);
        hR = __builtin_amdgcn_wmma_f32_16x16x32_f16(false, a, false, bRv, (short)0, hR, false, false);
        hZ = __builtin_amdgcn_wmma_f32_16x16x32_f16(false, a, false, bZv, (short)0, hZ, false, false);
        hN = __builtin_amdgcn_wmma_f32_16x16x32_f16(false, a, false, bNv, (short)0, hN, false, false);
      }
    }

    // register-resident gate math (C-layout aligns across r/z/n tiles)
    float hnew[8];
    for (int r8 = 0; r8 < 8; ++r8) {
      float rr = sigmoidf_((aR[r8] + bihR) + (hR[r8] + bhhR));
      float zz = sigmoidf_((aZ[r8] + bihZ) + (hZ[r8] + bhhZ));
      float nn = tanhf((aN[r8] + bihN) + rr * (hN[r8] + bhhN));
      float hold = Hs32[(rb + r8) * H_ + colR];
      hnew[r8] = (1.0f - zz) * nn + zz * hold;
    }
    __syncthreads();  // all waves done reading Hs16/Hs32
    for (int r8 = 0; r8 < 8; ++r8) {
      int m = rb + r8;
      Hs32[m * H_ + colR] = hnew[r8];
      Hs16[m * H_ + colR] = (half_t)hnew[r8];
      out16[((size_t)(bbase + m) * Tsteps + tact) * ostride + ocoff + colR] = (half_t)hnew[r8];
    }
  }
  __syncthreads();
  if (h_out32)
    for (int i = tid; i < BCH * H_; i += 512)
      h_out32[(size_t)(bbase + i / H_) * H_ + (i % H_)] = Hs32[i];
  if (h_out16)
    for (int i = tid; i < BCH * H_; i += 512)
      h_out16[(size_t)(bbase + i / H_) * hostride + ocoff + (i % H_)] = Hs16[i];
}

// ---------------------------------------------------------------------------
// Generic WMMA GEMM: C = act(A @ W^T + bias). A:(MxK) f16, W:(NxK) f16.
// block 128 = 4 waves; grid (M/16, N/64). act: 0 none, 1 relu, 2 tanh.
// trT>0: transposed f16 store for Eproj: C16[(m/trT)*ldc*trT + col*trT + m%trT]
// ---------------------------------------------------------------------------
__global__ __launch_bounds__(128) void wmma_gemm_kernel(
    const half_t* __restrict__ A, const half_t* __restrict__ W,
    const float* __restrict__ bias,
    float* __restrict__ C32, half_t* __restrict__ C16, int ldc,
    int K, int act, int trT) {
  __shared__ half_t As[16 * 512];  // K <= 512
  const int tid = threadIdx.x, wave = tid >> 5, lane = tid & 31;
  const int m0 = blockIdx.x * 16;
  const int n0 = blockIdx.y * 64 + wave * 16;
  for (int i = tid; i < 16 * K / 8; i += 128) {
    int m = i / (K / 8), kk = (i % (K / 8)) * 8;
    *(v8h*)&As[m * K + kk] = *(const v8h*)&A[(size_t)(m0 + m) * K + kk];
  }
  __syncthreads();
  const int c = lane & 15, hi = lane >> 4;
  const half_t* Wrow = W + (size_t)(n0 + c) * K;
  const half_t* arow = As + c * K;
  v8f acc = {};
  for (int k0 = 0; k0 < K; k0 += 32) {
    v16h a;
    ((v8h*)&a)[0] = *(const v8h*)(arow + k0 + hi * 16);
    ((v8h*)&a)[1] = *(const v8h*)(arow + k0 + hi * 16 + 8);
    v16h b = *(const v16h*)(Wrow + k0 + hi * 16);
    acc = __builtin_amdgcn_wmma_f32_16x16x32_f16(false, a, false, b, (short)0, acc, false, false);
  }
  const int col = n0 + c, rbase = hi * 8;
  const float bv = bias ? bias[col] : 0.0f;
  for (int r = 0; r < 8; ++r) {
    float v = acc[r] + bv;
    if (act == 1) v = fmaxf(v, 0.0f);
    else if (act == 2) v = tanhf(v);
    int m = m0 + rbase + r;
    if (trT > 0) {
      int bb = m / trT, tt = m % trT;
      C16[((size_t)bb * ldc + col) * trT + tt] = (half_t)v;
    } else {
      if (C32) C32[(size_t)m * ldc + col] = v;
      if (C16) C16[(size_t)m * ldc + col] = (half_t)v;
    }
  }
}

// ---------------------------------------------------------------------------
// Per-step attention: scores + softmax + context. One block per batch element.
// Ep layout (B, H, T) f16 so per-h iterations are lane-coalesced over tt.
// ---------------------------------------------------------------------------
__global__ __launch_bounds__(256) void attn_kernel(
    const float* __restrict__ hp, const half_t* __restrict__ Ep,
    const half_t* __restrict__ enc16, const float* __restrict__ vvec,
    float* __restrict__ attn_out, float* __restrict__ cbuf,
    half_t* __restrict__ xcat, int tstep) {
  __shared__ float hpS[H_];
  __shared__ float sc[PAST_];
  __shared__ float red[256];
  const int b = blockIdx.x, tid = threadIdx.x;
  hpS[tid] = hp[(size_t)b * H_ + tid];
  __syncthreads();
  float s0 = 0.0f, s1 = 0.0f;
  const half_t* Eb = Ep + (size_t)b * H_ * PAST_;
  for (int h = 0; h < H_; ++h) {
    float vh = vvec[h], hph = hpS[h];
    const half_t* row = Eb + (size_t)h * PAST_;
    s0 += vh * tanhf((float)row[tid] + hph);
    s1 += vh * tanhf((float)row[tid + 256] + hph);
  }
  sc[tid] = s0; sc[tid + 256] = s1;
  red[tid] = fmaxf(s0, s1);
  __syncthreads();
  for (int off = 128; off > 0; off >>= 1) {
    if (tid < off) red[tid] = fmaxf(red[tid], red[tid + off]);
    __syncthreads();
  }
  float mx = red[0];
  __syncthreads();
  float e0 = __expf(s0 - mx), e1 = __expf(s1 - mx);
  red[tid] = e0 + e1;
  __syncthreads();
  for (int off = 128; off > 0; off >>= 1) {
    if (tid < off) red[tid] += red[tid + off];
    __syncthreads();
  }
  float inv = 1.0f / red[0];
  float a0 = e0 * inv, a1 = e1 * inv;
  sc[tid] = a0; sc[tid + 256] = a1;
  attn_out[((size_t)b * PRED_ + tstep) * PAST_ + tid]       = a0;
  attn_out[((size_t)b * PRED_ + tstep) * PAST_ + tid + 256] = a1;
  __syncthreads();
  float c0 = 0.0f, c1 = 0.0f;
  const half_t* Ee = enc16 + (size_t)b * PAST_ * 512;
  for (int tt = 0; tt < PAST_; ++tt) {
    float av = sc[tt];
    c0 += av * (float)Ee[(size_t)tt * 512 + tid];
    c1 += av * (float)Ee[(size_t)tt * 512 + tid + 256];
  }
  cbuf[(size_t)b * 512 + tid] = c0;
  cbuf[(size_t)b * 512 + tid + 256] = c1;
  xcat[(size_t)b * 1024 + 256 + tid] = (half_t)c0;
  xcat[(size_t)b * 1024 + 256 + tid + 256] = (half_t)c1;
}

// fusion: q = relu(p_t @ fpW^T + fpb), u = relu(yprev * fyW + fyb)
__global__ __launch_bounds__(256) void fus_kernel(
    const float* __restrict__ p, int tstep, const float* __restrict__ yprev,
    const float* __restrict__ fpW, const float* __restrict__ fpb,
    const float* __restrict__ fyW, const float* __restrict__ fyb,
    half_t* __restrict__ xcat, float* __restrict__ ubuf) {
  const int b = blockIdx.x, h = threadIdx.x;
  const float* pt = p + ((size_t)b * PRED_ + tstep) * 8;
  float q = fpb[h];
  for (int j = 0; j < 8; ++j) q += pt[j] * fpW[h * 8 + j];
  q = fmaxf(q, 0.0f);
  float u = fmaxf(yprev[b] * fyW[h] + fyb[h], 0.0f);
  xcat[(size_t)b * 1024 + h]       = (half_t)q;
  xcat[(size_t)b * 1024 + 768 + h] = (half_t)u;
  ubuf[(size_t)b * H_ + h] = u;
}

// y = [h1 | c | u] @ out_W + out_b  (one block per batch element)
__global__ __launch_bounds__(256) void outgemv_kernel(
    const float* __restrict__ h1, const float* __restrict__ cbuf,
    const float* __restrict__ ubuf, const float* __restrict__ outW,
    const float* __restrict__ outb, float* __restrict__ ypred,
    float* __restrict__ yprev, int tstep) {
  __shared__ float red[256];
  const int b = blockIdx.x, i = threadIdx.x;
  float s = outW[i] * h1[(size_t)b * H_ + i]
          + outW[256 + i] * cbuf[(size_t)b * 512 + i]
          + outW[512 + i] * cbuf[(size_t)b * 512 + 256 + i]
          + outW[768 + i] * ubuf[(size_t)b * H_ + i];
  red[i] = s;
  __syncthreads();
  for (int off = 128; off > 0; off >>= 1) {
    if (i < off) red[i] += red[i + off];
    __syncthreads();
  }
  if (i == 0) {
    float y = red[0] + outb[0];
    ypred[(size_t)b * PRED_ + tstep] = y;
    yprev[b] = y;
  }
}

// ---------------------------------------------------------------------------
// Host orchestration
// ---------------------------------------------------------------------------
extern "C" void kernel_launch(void* const* d_in, const int* in_sizes, int n_in,
                              void* d_out, int out_size, void* d_ws, size_t ws_size,
                              hipStream_t stream) {
  (void)in_sizes; (void)n_in; (void)out_size; (void)ws_size;
  const float* x_past = (const float*)d_in[0];
  const float* p      = (const float*)d_in[1];
  const float* fcW    = (const float*)d_in[3];
  const float* fcb    = (const float*)d_in[4];
  const float *eW[2][2], *eU[2][2], *ebi[2][2], *ebh[2][2];
  int idx = 5;
  for (int l = 0; l < 2; ++l)
    for (int d = 0; d < 2; ++d) {
      eW[l][d]  = (const float*)d_in[idx++];
      eU[l][d]  = (const float*)d_in[idx++];
      ebi[l][d] = (const float*)d_in[idx++];
      ebh[l][d] = (const float*)d_in[idx++];
    }
  const float* tr_W   = (const float*)d_in[21];
  const float* tr_b   = (const float*)d_in[22];
  const float* attn_W = (const float*)d_in[23];
  const float* attn_v = (const float*)d_in[24];
  const float* fpW = (const float*)d_in[25];
  const float* fpb = (const float*)d_in[26];
  const float* fyW = (const float*)d_in[27];
  const float* fyb = (const float*)d_in[28];
  const float *dW[2], *dU[2], *dbi[2], *dbh[2];
  for (int l = 0; l < 2; ++l) {
    dW[l]  = (const float*)d_in[29 + 4 * l];
    dU[l]  = (const float*)d_in[30 + 4 * l];
    dbi[l] = (const float*)d_in[31 + 4 * l];
    dbh[l] = (const float*)d_in[32 + 4 * l];
  }
  const float* out_W = (const float*)d_in[37];
  const float* out_b = (const float*)d_in[38];

  float* ypred    = (float*)d_out;
  float* attn_out = (float*)d_out + (size_t)B_ * PRED_;

  uintptr_t base = (uintptr_t)d_ws;
  size_t off = 0;
  auto carve = [&](size_t bytes) -> void* {
    void* ptr = (void*)(base + off);
    off = (off + bytes + 255) & ~(size_t)255;
    return ptr;
  };
  half_t* Xbuf  = (half_t*)carve((size_t)B_ * PAST_ * H_ * 2);        // 67 MB
  half_t* L1in  = (half_t*)carve((size_t)B_ * PAST_ * 2 * H_ * 2);    // 134 MB
  half_t* enc16 = (half_t*)carve((size_t)B_ * PAST_ * 2 * H_ * 2);    // 134 MB
  half_t* Ep    = (half_t*)carve((size_t)B_ * H_ * PAST_ * 2);        // 67 MB (B,H,T)
  half_t* Wenc[2][2];
  Wenc[0][0] = (half_t*)carve((size_t)G_ * 512 * 2);
  Wenc[0][1] = (half_t*)carve((size_t)G_ * 512 * 2);
  Wenc[1][0] = (half_t*)carve((size_t)G_ * 768 * 2);
  Wenc[1][1] = (half_t*)carve((size_t)G_ * 768 * 2);
  half_t* Wdec0 = (half_t*)carve((size_t)G_ * 1280 * 2);
  half_t* Wdec1 = (half_t*)carve((size_t)G_ * 512 * 2);
  half_t* Wtr   = (half_t*)carve((size_t)H_ * 512 * 2);
  half_t* Wq    = (half_t*)carve((size_t)H_ * H_ * 2);
  half_t* We    = (half_t*)carve((size_t)H_ * 512 * 2);
  half_t* hcat  = (half_t*)carve((size_t)2 * B_ * 512 * 2);
  float*  hdec  = (float*)carve((size_t)2 * B_ * H_ * 4);
  half_t* hdec16= (half_t*)carve((size_t)2 * B_ * H_ * 2);
  float*  hp    = (float*)carve((size_t)B_ * H_ * 4);
  float*  cbuf  = (float*)carve((size_t)B_ * 512 * 4);
  float*  ubuf  = (float*)carve((size_t)B_ * H_ * 4);
  half_t* xcat  = (half_t*)carve((size_t)B_ * 1024 * 2);
  float*  yprev = (float*)carve((size_t)B_ * 4);

  auto cdiv = [](size_t a, size_t b) { return (int)((a + b - 1) / b); };

  // ---- weight packing (f16) ----
  gruw_pack_kernel<<<cdiv((size_t)G_ * 512, 256), 256, 0, stream>>>(eW[0][0], eU[0][0], Wenc[0][0], 256);
  gruw_pack_kernel<<<cdiv((size_t)G_ * 512, 256), 256, 0, stream>>>(eW[0][1], eU[0][1], Wenc[0][1], 256);
  gruw_pack_kernel<<<cdiv((size_t)G_ * 768, 256), 256, 0, stream>>>(eW[1][0], eU[1][0], Wenc[1][0], 512);
  gruw_pack_kernel<<<cdiv((size_t)G_ * 768, 256), 256, 0, stream>>>(eW[1][1], eU[1][1], Wenc[1][1], 512);
  gruw_pack_kernel<<<cdiv((size_t)G_ * 1280, 256), 256, 0, stream>>>(dW[0], dU[0], Wdec0, 1024);
  gruw_pack_kernel<<<cdiv((size_t)G_ * 512, 256), 256, 0, stream>>>(dW[1], dU[1], Wdec1, 256);
  cast_f16_kernel<<<cdiv((size_t)H_ * 512, 256), 256, 0, stream>>>(tr_W, Wtr, H_ * 512);
  attnw_split_kernel<<<cdiv((size_t)H_ * G_, 256), 256, 0, stream>>>(attn_W, Wq, We);

  // ---- encoder ----
  encfc_kernel<<<cdiv((size_t)B_ * PAST_ * H_, 256), 256, 0, stream>>>(x_past, fcW, fcb, Xbuf);
  y0_kernel<<<1, 256, 0, stream>>>(x_past, yprev);

  // layer 0 (din=256), both directions in one launch via gridDim.y
  gru_scan_kernel<<<dim3(B_ / BCH, 2), 512, 0, stream>>>(
      Xbuf, 256, Wenc[0][0], Wenc[0][1], ebi[0][0], ebi[0][1], ebh[0][0], ebh[0][1],
      nullptr, nullptr, hcat, 512, L1in, 512, PAST_, 1);
  // layer 1 (din=512)
  gru_scan_kernel<<<dim3(B_ / BCH, 2), 512, 0, stream>>>(
      L1in, 512, Wenc[1][0], Wenc[1][1], ebi[1][0], ebi[1][1], ebh[1][0], ebh[1][1],
      nullptr, nullptr, hcat + (size_t)B_ * 512, 512, enc16, 512, PAST_, 1);

  // decoder initial h: tanh([hf|hb] @ tr_W^T + tr_b) for both layers stacked (M=512)
  wmma_gemm_kernel<<<dim3(2 * B_ / 16, H_ / 64), 128, 0, stream>>>(
      hcat, Wtr, tr_b, hdec, hdec16, H_, 512, /*act=*/2, /*trT=*/0);

  // Eproj = enc_out @ We^T, stored transposed (B,H,T) f16
  wmma_gemm_kernel<<<dim3((size_t)B_ * PAST_ / 16, H_ / 64), 128, 0, stream>>>(
      enc16, We, nullptr, nullptr, Ep, H_, 512, /*act=*/0, /*trT=*/PAST_);

  // ---- decoder: 48 steps ----
  for (int t = 0; t < PRED_; ++t) {
    // hp = hq @ Wq^T (hq = layer-1 hidden, f16)
    wmma_gemm_kernel<<<dim3(B_ / 16, H_ / 64), 128, 0, stream>>>(
        hdec16 + (size_t)B_ * H_, Wq, nullptr, hp, nullptr, H_, H_, 0, 0);
    attn_kernel<<<B_, 256, 0, stream>>>(hp, Ep, enc16, attn_v, attn_out, cbuf, xcat, t);
    fus_kernel<<<B_, 256, 0, stream>>>(p, t, yprev, fpW, fpb, fyW, fyb, xcat, ubuf);
    // decoder GRU layer 0 (din=1024): x = [q|c|u]
    gru_scan_kernel<<<dim3(B_ / BCH, 1), 512, 0, stream>>>(
        xcat, 1024, Wdec0, Wdec0, dbi[0], dbi[0], dbh[0], dbh[0],
        hdec, hdec, nullptr, H_, hdec16, H_, 1, 0);
    // decoder GRU layer 1 (din=256): x = h0
    gru_scan_kernel<<<dim3(B_ / BCH, 1), 512, 0, stream>>>(
        hdec16, 256, Wdec1, Wdec1, dbi[1], dbi[1], dbh[1], dbh[1],
        hdec + (size_t)B_ * H_, hdec + (size_t)B_ * H_, nullptr, H_,
        hdec16 + (size_t)B_ * H_, H_, 1, 0);
    outgemv_kernel<<<B_, 256, 0, stream>>>(hdec + (size_t)B_ * H_, cbuf, ubuf,
                                           out_W, out_b, ypred, yprev, t);
  }
}
